// CausalSelfAttention_77695958385270
// MI455X (gfx1250) — matrix operand order repaired and checked by reading, not compiled, not run on hardware
//
#include <hip/hip_runtime.h>
#include <hip/hip_bf16.h>
#include <math.h>

// ---------------------------------------------------------------------------
// CausalSelfAttention forward for MI455X (gfx1250, wave32, WMMA).
// B=1, T=2048, DIM=2048, NH=32, NKV=8, HD=64, REP=4.
// GEMMs: LDS block-tiled (128x64, BK=64), double-buffered async copies
//        (global_load_async_to_lds_b128 + s_wait_asynccnt), 32x64 wave tiles,
//        v_wmma_f32_16x16x32_bf16.
// Attention: per-wave flash with online softmax, WMMA bf16.
// ---------------------------------------------------------------------------

typedef __bf16 bf16_t;
typedef __attribute__((ext_vector_type(16))) __bf16 v16bf;
typedef __attribute__((ext_vector_type(8)))  __bf16 v8bf;
typedef __attribute__((ext_vector_type(8)))  float  v8f;

#define T_SEQ    2048
#define DIM_C    2048
#define NH_C     32
#define NKV_C    8
#define HD_C     64
#define REP_C    (NH_C / NKV_C)
#define QKV_COLS ((NH_C + 2 * NKV_C) * HD_C)   // 3072

union bfu { v16bf v; v8bf h[2]; };

__device__ __forceinline__ v8f wmma_bf16(v16bf a, v16bf b, v8f c) {
  // D = A(16x32 bf16) * B(32x16 bf16) + C(16x16 f32)
  return __builtin_amdgcn_wmma_f32_16x16x32_bf16(false, a, false, b,
                                                 (short)0, c, false, false);
}

// ---- async global->LDS copy (ASYNCcnt-tracked), with sync fallback --------
#if defined(__has_builtin)
#if __has_builtin(__builtin_amdgcn_global_load_async_to_lds_b128)
#define HAVE_ASYNC_LDS 1
#endif
#endif
#ifndef HAVE_ASYNC_LDS
#define HAVE_ASYNC_LDS 0
#endif

typedef int v4i __attribute__((vector_size(4 * sizeof(int))));
typedef __attribute__((address_space(3))) v4i* as3_v4i;

__device__ __forceinline__ void cp16_g2l(const bf16_t* g, bf16_t* l) {
#if HAVE_ASYNC_LDS
  __builtin_amdgcn_global_load_async_to_lds_b128((v4i*)g, (as3_v4i)l, 0, 0);
#else
  *(v8bf*)l = *(const v8bf*)g;
#endif
}

#if HAVE_ASYNC_LDS
#define WAIT_ASYNC_PREV() asm volatile("s_wait_asynccnt 0xc" ::: "memory")
#define WAIT_ASYNC_ALL()  asm volatile("s_wait_asynccnt 0x0" ::: "memory")
#else
#define WAIT_ASYNC_PREV() do {} while (0)
#define WAIT_ASYNC_ALL()  do {} while (0)
#endif

// ---------------------------------------------------------------------------
// Kernel 1: f32 -> bf16 convert
// ---------------------------------------------------------------------------
__global__ void cvt_f32_to_bf16(const float* __restrict__ in,
                                bf16_t* __restrict__ out, int n) {
  int i = blockIdx.x * blockDim.x + threadIdx.x;
  if (i < n) out[i] = (bf16_t)in[i];
}

// ---------------------------------------------------------------------------
// Kernel 2/5: C[M,N] = A[M,K] * B[N,K]^T, bf16 in, WMMA f32 accumulate.
// Block: 128 threads (4 waves). Block tile: 128(M) x 64(N), BK = 64.
// LDS: A[2][128][64] (32KB) + B[2][64][64] (16KB), double buffered (48KB).
// Wave w computes rows [w*32, w*32+32) x all 64 N cols: 2 A-frags x 4 B-frags
// -> 8 WMMAs per k32 step, 8 f32 accumulators (each B-frag feeds 2 WMMAs).
// Fragment layouts (A/B use matching contiguous k, not the ISA 7.12.2 table order):
//   A-frag 16x32: lane=row, half e -> K=16*hi+e
//   B-frag 32x16: lane=col, half e -> K=16*hi+e
//   C-frag 16x16: vgpr r -> row=r+8*hi, col=lane&15
// ---------------------------------------------------------------------------
template <bool OUT_F32>
__global__ __launch_bounds__(128)
void gemm_bf16_wmma(const bf16_t* __restrict__ A, const bf16_t* __restrict__ B,
                    void* __restrict__ C, int M, int N, int K) {
  __shared__ __align__(32) bf16_t As[2][128][64];
  __shared__ __align__(32) bf16_t Bs[2][64][64];

  const int tid  = threadIdx.x;
  const int lane = tid & 31;
  const int wib  = tid >> 5;                 // wave in block: 0..3
  const int ntn  = N >> 6;                   // N tiles of 64
  const int m0   = (blockIdx.x / ntn) << 7;  // block M origin (x128)
  const int n0   = (blockIdx.x % ntn) << 6;  // block N origin (x64)
  const int hi = lane >> 4;
  const int lo = lane & 15;

  // ---- async tile fetch: A = 1024 16B chunks, B = 512 16B chunks ----------
  auto issue_tile = [&](int buf, int k0) {
#pragma unroll
    for (int i = 0; i < 8; ++i) {            // A tile: 128 rows x 128B
      const int c = tid + i * 128;
      const int row = c >> 3, cc = c & 7;
      cp16_g2l(A + (size_t)(m0 + row) * K + k0 + cc * 8,
               &As[buf][row][cc * 8]);
    }
#pragma unroll
    for (int i = 0; i < 4; ++i) {            // B tile: 64 rows x 128B
      const int c = tid + i * 128;
      const int row = c >> 3, cc = c & 7;
      cp16_g2l(B + (size_t)(n0 + row) * K + k0 + cc * 8,
               &Bs[buf][row][cc * 8]);
    }
  };

  v8f acc[2][4] = {};
  const int wm = wib << 5;                   // wave M offset inside tile (x32)
  const int nk = K >> 6;                     // BK = 64 steps

  issue_tile(0, 0);
  int buf = 0;
  for (int ks = 0; ks < nk; ++ks) {
    if (ks + 1 < nk) {
      issue_tile(buf ^ 1, (ks + 1) << 6);
      WAIT_ASYNC_PREV();                     // oldest 12 copies (this buf) done
    } else {
      WAIT_ASYNC_ALL();
    }
    __syncthreads();                         // tile visible to all waves

    const bf16_t* As_ = &As[buf][0][0];
    const bf16_t* Bs_ = &Bs[buf][0][0];
#pragma unroll
    for (int k32 = 0; k32 < 64; k32 += 32) {
      bfu a0, a1;
      a0.h[0] = *(const v8bf*)(As_ + (wm + lo) * 64 + k32 + 16 * hi);
      a0.h[1] = *(const v8bf*)(As_ + (wm + lo) * 64 + k32 + 8 + 16 * hi);
      a1.h[0] = *(const v8bf*)(As_ + (wm + 16 + lo) * 64 + k32 + 16 * hi);
      a1.h[1] = *(const v8bf*)(As_ + (wm + 16 + lo) * 64 + k32 + 8 + 16 * hi);
#pragma unroll
      for (int j = 0; j < 4; ++j) {
        v16bf b = *(const v16bf*)(Bs_ + (j * 16 + lo) * 64 + k32 + 16 * hi);
        acc[0][j] = wmma_bf16(a0.v, b, acc[0][j]);
        acc[1][j] = wmma_bf16(a1.v, b, acc[1][j]);
      }
    }
    __syncthreads();                         // all reads done before overwrite
    buf ^= 1;
  }

#pragma unroll
  for (int rr = 0; rr < 2; ++rr) {
#pragma unroll
    for (int j = 0; j < 4; ++j) {
#pragma unroll
      for (int r = 0; r < 8; ++r) {
        const size_t row = (size_t)(m0 + wm + rr * 16 + r + 8 * hi);
        const size_t col = (size_t)(n0 + j * 16 + lo);
        if (OUT_F32) ((float*)C)[row * N + col] = acc[rr][j][r];
        else         ((bf16_t*)C)[row * N + col] = (bf16_t)acc[rr][j][r];
      }
    }
  }
}

// ---------------------------------------------------------------------------
// Kernel 3: split qkv + RoPE (+q gain/temp/softmax-scale) + V transpose.
//   q  -> [NH][T][HD]   (RoPE'd, scaled by q_gain*attn_temp/sqrt(HD))
//   k  -> [NKV][T][HD]  (RoPE'd)
//   vt -> [NKV][HD][T]  (transposed for contiguous PV B-fragments)
// ---------------------------------------------------------------------------
__global__ void rope_split(const bf16_t* __restrict__ qkv,
                           const float* __restrict__ q_gain,
                           const float* __restrict__ attn_temp,
                           bf16_t* __restrict__ qd, bf16_t* __restrict__ kd,
                           bf16_t* __restrict__ vt) {
  const int gid = blockIdx.x * blockDim.x + threadIdx.x;
  const int d  = gid & 31;
  const int t  = (gid >> 5) & (T_SEQ - 1);
  const int hp = gid >> 16;                       // / (32 * 2048)
  if (hp >= NH_C + 2 * NKV_C) return;

  const bf16_t* row = qkv + (size_t)t * QKV_COLS + hp * HD_C;
  const float x1 = (float)row[d];
  const float x2 = (float)row[d + 32];

  if (hp < NH_C + NKV_C) {
    const float ang = (float)t * ((float)d * (1.0f / 32.0f));
    const float s = sinf(ang), c = cosf(ang);
    float o1 = x1 * c - x2 * s;
    float o2 = x1 * s + x2 * c;
    if (hp < NH_C) {
      const float g = q_gain[hp] * attn_temp[hp] * 0.125f;  // 1/sqrt(64)
      o1 *= g; o2 *= g;
      bf16_t* out = qd + ((size_t)hp * T_SEQ + t) * HD_C;
      out[d] = (bf16_t)o1; out[d + 32] = (bf16_t)o2;
    } else {
      bf16_t* out = kd + ((size_t)(hp - NH_C) * T_SEQ + t) * HD_C;
      out[d] = (bf16_t)o1; out[d + 32] = (bf16_t)o2;
    }
  } else {
    bf16_t* out = vt + (size_t)(hp - NH_C - NKV_C) * HD_C * T_SEQ;
    out[(size_t)d * T_SEQ + t]        = (bf16_t)x1;
    out[(size_t)(d + 32) * T_SEQ + t] = (bf16_t)x2;
  }
}

// ---------------------------------------------------------------------------
// Kernel 4: causal flash attention. 1 wave = 16 query rows; 4 waves/block.
// Per kv step of 32 keys: S(16x32) via 4 WMMAs, online softmax (shfl_xor row
// reduce over 16-lane halves), P staged through per-wave LDS to re-fragment
// C-layout -> A-layout (s_wait_dscnt), then O += P*V via 4 WMMAs.
// ---------------------------------------------------------------------------
__global__ __launch_bounds__(128)
void attn_fwd(const bf16_t* __restrict__ q, const bf16_t* __restrict__ k,
              const bf16_t* __restrict__ vt, bf16_t* __restrict__ y) {
  __shared__ __align__(16) bf16_t pshare[4][16 * 32];

  const int lane = threadIdx.x & 31;
  const int w    = threadIdx.x >> 5;
  const int qblocks = T_SEQ / 64;               // 32
  const int head = blockIdx.x / qblocks;
  const int q0   = (blockIdx.x % qblocks) * 64 + w * 16;
  const int kvh  = head / REP_C;
  const int hi = lane >> 4, lo = lane & 15;

  const bf16_t* qbase = q  + ((size_t)head * T_SEQ + q0) * HD_C;
  const bf16_t* kbase = k  + (size_t)kvh * T_SEQ * HD_C;
  const bf16_t* vbase = vt + (size_t)kvh * HD_C * T_SEQ;

  bfu aq0, aq1;
  {
    const bf16_t* qrow = qbase + lo * HD_C;
    aq0.h[0] = *(const v8bf*)(qrow +  0 + 16 * hi);
    aq0.h[1] = *(const v8bf*)(qrow +  8 + 16 * hi);
    aq1.h[0] = *(const v8bf*)(qrow + 32 + 16 * hi);
    aq1.h[1] = *(const v8bf*)(qrow + 40 + 16 * hi);
  }

  v8f o[4] = {};
  float mrow[8], lrow[8];
#pragma unroll
  for (int r = 0; r < 8; ++r) { mrow[r] = -__builtin_inff(); lrow[r] = 0.0f; }

  bf16_t* plds = &pshare[w][0];
  const int qmax = q0 + 15;

  for (int kv0 = 0; kv0 <= qmax; kv0 += 32) {
    v8f s0 = {}, s1 = {};
    {
      v16bf b0 = *(const v16bf*)(kbase + (size_t)(kv0 + lo) * HD_C + 16 * hi);
      v16bf b1 = *(const v16bf*)(kbase + (size_t)(kv0 + lo) * HD_C + 32 + 16 * hi);
      s0 = wmma_bf16(aq0.v, b0, s0);
      s0 = wmma_bf16(aq1.v, b1, s0);
      v16bf b2 = *(const v16bf*)(kbase + (size_t)(kv0 + 16 + lo) * HD_C + 16 * hi);
      v16bf b3 = *(const v16bf*)(kbase + (size_t)(kv0 + 16 + lo) * HD_C + 32 + 16 * hi);
      s1 = wmma_bf16(aq0.v, b2, s1);
      s1 = wmma_bf16(aq1.v, b3, s1);
    }

    const int col0 = kv0 + lo, col1 = kv0 + 16 + lo;
#pragma unroll
    for (int r = 0; r < 8; ++r) {
      const int row = q0 + r + 8 * hi;
      float a0 = (col0 <= row) ? s0[r] : -__builtin_inff();
      float a1 = (col1 <= row) ? s1[r] : -__builtin_inff();
      float mx = fmaxf(a0, a1);
      mx = fmaxf(mx, __shfl_xor(mx, 1));
      mx = fmaxf(mx, __shfl_xor(mx, 2));
      mx = fmaxf(mx, __shfl_xor(mx, 4));
      mx = fmaxf(mx, __shfl_xor(mx, 8));
      const float mnew = fmaxf(mrow[r], mx);
      const float p0 = __expf(a0 - mnew);
      const float p1 = __expf(a1 - mnew);
      float ps = p0 + p1;
      ps += __shfl_xor(ps, 1);
      ps += __shfl_xor(ps, 2);
      ps += __shfl_xor(ps, 4);
      ps += __shfl_xor(ps, 8);
      const float alpha = __expf(mrow[r] - mnew);
      lrow[r] = lrow[r] * alpha + ps;
      mrow[r] = mnew;
#pragma unroll
      for (int j = 0; j < 4; ++j) o[j][r] *= alpha;
      plds[(r + 8 * hi) * 32 + lo]      = (bf16_t)p0;
      plds[(r + 8 * hi) * 32 + 16 + lo] = (bf16_t)p1;
    }

    asm volatile("s_wait_dscnt 0x0" ::: "memory");

    bfu ap;
    ap.h[0] = *(const v8bf*)(plds + lo * 32 + 16 * hi);
    ap.h[1] = *(const v8bf*)(plds + lo * 32 + 8 + 16 * hi);
#pragma unroll
    for (int j = 0; j < 4; ++j) {
      v16bf bv = *(const v16bf*)(vbase + (size_t)(j * 16 + lo) * T_SEQ + kv0 + 16 * hi);
      o[j] = wmma_bf16(ap.v, bv, o[j]);
    }
  }

#pragma unroll
  for (int j = 0; j < 4; ++j) {
#pragma unroll
    for (int r = 0; r < 8; ++r) {
      const int row = q0 + r + 8 * hi;
      const float val = o[j][r] / lrow[r];
      y[(size_t)row * DIM_C + head * HD_C + j * 16 + lo] = (bf16_t)val;
    }
  }
}

// ---------------------------------------------------------------------------
// Host-side orchestration
// ---------------------------------------------------------------------------
extern "C" void kernel_launch(void* const* d_in, const int* in_sizes, int n_in,
                              void* d_out, int out_size, void* d_ws, size_t ws_size,
                              hipStream_t stream) {
  (void)in_sizes; (void)n_in; (void)out_size; (void)ws_size;

  const float* x         = (const float*)d_in[0];
  const float* Wqkv      = (const float*)d_in[1];
  const float* Wproj     = (const float*)d_in[2];
  const float* q_gain    = (const float*)d_in[3];
  const float* attn_temp = (const float*)d_in[4];

  char* ws = (char*)d_ws;
  bf16_t* xb     = (bf16_t*)ws; ws += (size_t)T_SEQ * DIM_C * 2;
  bf16_t* wqkvb  = (bf16_t*)ws; ws += (size_t)QKV_COLS * DIM_C * 2;
  bf16_t* wprojb = (bf16_t*)ws; ws += (size_t)DIM_C * DIM_C * 2;
  bf16_t* qkvb   = (bf16_t*)ws; ws += (size_t)T_SEQ * QKV_COLS * 2;
  bf16_t* qd     = (bf16_t*)ws; ws += (size_t)NH_C * T_SEQ * HD_C * 2;
  bf16_t* kd     = (bf16_t*)ws; ws += (size_t)NKV_C * T_SEQ * HD_C * 2;
  bf16_t* vtd    = (bf16_t*)ws; ws += (size_t)NKV_C * HD_C * T_SEQ * 2;
  bf16_t* yb     = (bf16_t*)ws; ws += (size_t)T_SEQ * DIM_C * 2;

  const int nx = T_SEQ * DIM_C;
  const int nq = QKV_COLS * DIM_C;
  const int np = DIM_C * DIM_C;
  cvt_f32_to_bf16<<<(nx + 255) / 256, 256, 0, stream>>>(x, xb, nx);
  cvt_f32_to_bf16<<<(nq + 255) / 256, 256, 0, stream>>>(Wqkv, wqkvb, nq);
  cvt_f32_to_bf16<<<(np + 255) / 256, 256, 0, stream>>>(Wproj, wprojb, np);

  // qkv = x @ Wqkv^T : M=2048, N=3072, K=2048 -> (16 x 48) = 768 blocks
  gemm_bf16_wmma<false><<<(T_SEQ / 128) * (QKV_COLS / 64), 128, 0, stream>>>(
      xb, wqkvb, qkvb, T_SEQ, QKV_COLS, DIM_C);

  // RoPE + split
  {
    const int n = (NH_C + 2 * NKV_C) * T_SEQ * 32;
    rope_split<<<n / 256, 256, 0, stream>>>(qkvb, q_gain, attn_temp, qd, kd, vtd);
  }

  // flash attention: 32 heads * 32 query-blocks of 64, 128 threads (4 waves)
  attn_fwd<<<NH_C * (T_SEQ / 64), 128, 0, stream>>>(qd, kd, vtd, yb);

  // out = y @ Wproj^T : M=2048, N=2048, K=2048 -> (16 x 32) = 512 blocks
  gemm_bf16_wmma<true><<<(T_SEQ / 128) * (DIM_C / 64), 128, 0, stream>>>(
      yb, wprojb, d_out, T_SEQ, DIM_C, DIM_C);
}
